// GNNModel_48945447305999
// MI455X (gfx1250) — matrix operand order repaired and checked by reading, not compiled, not run on hardware
//
#include <hip/hip_runtime.h>
#include <hip/hip_bf16.h>

// Fully fused sudoku-GCN forward for MI455X (gfx1250, wave32, WMMA).
// h stays resident in LDS across all 6 GCN layers; all GEMMs use
// v_wmma_f32_16x16x32_bf16 (bf16 inputs, fp32 accumulate) with 2x2
// register blocking (32x32 macro-tile per wave -> 1 LDS frag load/WMMA).

typedef __bf16 bf16_t;
typedef __attribute__((ext_vector_type(16))) __bf16        v16bf;
typedef __attribute__((ext_vector_type(8)))  float         v8f;
typedef __attribute__((ext_vector_type(4)))  unsigned int  v4u;

constexpr int BT    = 2;     // batch elements per workgroup
constexpr int NODES = 81;
constexpr int NPAD  = 96;    // 6 tiles of 16
constexpr int H     = 128;
constexpr int NLAY  = 6;
constexpr int HS    = 136;   // h row stride (bf16 elems): 272B/row -> conflict-free frags
constexpr int TS    = 104;   // T1^T row stride
constexpr int AS    = 104;   // A row stride
constexpr int WSR   = 136;   // W^T row stride

constexpr int H_ELEMS = BT * NPAD * HS;   // 26112
constexpr int T_ELEMS = BT * H * TS;      // 26624
constexpr int A_ELEMS = NPAD * AS;        //  9984
constexpr int W_ELEMS = H * WSR;          // 17408
constexpr size_t SMEM_BYTES =
    (size_t)(H_ELEMS + T_ELEMS + A_ELEMS + W_ELEMS) * sizeof(bf16_t) + H * sizeof(float);

// A-matrix fragment (16x32, 16-bit): per ISA 7.12.2 — lane lo=M row;
// elems 0..7 = K kb..kb+7, elems 8..15 = K kb+16..kb+23 (kb = k0 + 8*hi).
__device__ inline v16bf load_frag_a(const bf16_t* base16B) {
  union { v4u q[2]; v16bf v; } u;
  u.q[0] = *(const v4u*)(base16B);
  u.q[1] = *(const v4u*)(base16B + 16);
  return u.v;
}
// B-matrix fragment (32x16, 16-bit): lane lo=N col; elems 0..7 = K kb..kb+7,
// elems 8..15 = K kb+16..kb+23 (kb = k0 + 8*hi).
__device__ inline v16bf load_frag_b(const bf16_t* base16B) {
  union { v4u q[2]; v16bf v; } u;
  u.q[0] = *(const v4u*)(base16B);
  u.q[1] = *(const v4u*)(base16B + 16);
  return u.v;
}

__device__ inline v8f wmma_bf16(v16bf a, v16bf b, v8f c) {
  return __builtin_amdgcn_wmma_f32_16x16x32_bf16(false, a, false, b,
                                                 (short)0, c, false, false);
}

// D elem v of a 16x16 f32 tile: M = m0+v+8*hi, N = n0+lo.
__device__ inline void store_t1_tile(bf16_t* t1base, const v8f& acc,
                                     int n0, int m0, int lo, int hi) {
  bf16_t* p = t1base + (n0 + lo) * TS + m0 + hi * 8;
#pragma unroll
  for (int v = 0; v < 8; ++v) p[v] = (bf16_t)acc[v];   // contiguous -> b128 store
}

__device__ inline void store_h_tile(bf16_t* hbase, const v8f& acc,
                                    int m0, int n0, float bias, int lo, int hi) {
  bf16_t* p = hbase + (m0 + hi * 8) * HS + (n0 + lo);
#pragma unroll
  for (int v = 0; v < 8; ++v) {
    float vv = acc[v] + bias;
    vv = vv > 0.0f ? vv : 0.0f;
    p[v * HS] = (bf16_t)vv;
  }
}

__global__ __launch_bounds__(256) void sudoku_gcn_fused(
    const int*   __restrict__ x,        // [B,81] digits 0..9
    const float* __restrict__ W_in,     // [10,128]
    const float* __restrict__ b_in,     // [128]
    const float* __restrict__ W_layers, // [6,128,128]
    const float* __restrict__ b_layers, // [6,128]
    const float* __restrict__ W_out,    // [128,9]
    const float* __restrict__ b_out,    // [9]
    float*       __restrict__ out)      // [B,81,9]
{
  extern __shared__ char smem_raw[];
  bf16_t* hb    = (bf16_t*)smem_raw;            // [BT][NPAD][HS]
  bf16_t* t1b   = hb  + H_ELEMS;                // [BT][H][TS]  (T1 transposed)
  bf16_t* Ab    = t1b + T_ELEMS;                // [NPAD][AS]
  bf16_t* Wb    = Ab  + A_ELEMS;                // [H][WSR]     (W transposed)
  float*  biasb = (float*)(Wb + W_ELEMS);       // [H]

  const int tid  = threadIdx.x;
  const int lane = tid & 31, wave = tid >> 5;
  const int lo   = lane & 15, hi = lane >> 4;
  const int b0   = blockIdx.x * BT;

  // ---- build D^{-1/2}(A+I)D^{-1/2}: every sudoku node has degree 21 -> A/21
  for (int t = tid; t < NPAD * AS; t += 256) {
    int r = t / AS, c = t - r * AS;
    bf16_t v = (bf16_t)0.0f;
    if (r < NODES && c < NODES) {
      int r1 = r / 9, c1 = r - r1 * 9, r2 = c / 9, c2 = c - r2 * 9;
      bool rel = (r1 == r2) | (c1 == c2) | ((r1 / 3 == r2 / 3) & (c1 / 3 == c2 / 3));
      if (rel) v = (bf16_t)(1.0f / 21.0f);
    }
    Ab[t] = v;
  }

  // ---- input projection: one-hot matmul == row gather of W_in, +b, relu
  for (int t = tid; t < BT * NPAD * H; t += 256) {
    int bi = t / (NPAD * H);
    int r  = t - bi * NPAD * H;
    int node = r / H, ch = r - node * H;
    float v = 0.0f;
    if (node < NODES) {
      int digit = x[(b0 + bi) * NODES + node];
      v = W_in[digit * H + ch] + b_in[ch];
      v = v > 0.0f ? v : 0.0f;
    }
    hb[bi * (NPAD * HS) + node * HS + ch] = (bf16_t)v;
  }
  __syncthreads();

  for (int l = 0; l < NLAY; ++l) {
    // stage W_l transposed (bf16) + bias into LDS
    const float* Wl = W_layers + (size_t)l * H * H;
    for (int t = tid; t < H * H; t += 256) {
      int c = t / H, n = t - c * H;          // Wl row-major [c][n]
      Wb[n * WSR + c] = (bf16_t)Wl[t];
    }
    if (tid < H) biasb[tid] = b_layers[l * H + tid];
    if (l + 1 < NLAY) {                       // warm L2 with next layer's 64KB
      const char* nxt = (const char*)(W_layers + (size_t)(l + 1) * H * H);
      __builtin_prefetch(nxt + tid * 256, 0, 1);
    }
    __syncthreads();

    // GEMM1: T1^T[ch][node] = sum_c h[node][c] * W[c][ch]  (M=96,N=128,K=128)
    // 2x2 register blocking: 24 macro-tiles (BT*3*4) over 8 waves = 3 iters.
#pragma unroll
    for (int it = 0; it < 3; ++it) {
      int t  = it * 8 + wave;                 // 0..23, wave-uniform
      int bi = t / 12, r = t - bi * 12;
      int m0 = (r >> 2) * 32, n0 = (r & 3) * 32;
      const bf16_t* hrow0 = hb + bi * (NPAD * HS) + (m0 + lo) * HS;
      const bf16_t* hrow1 = hrow0 + 16 * HS;
      const bf16_t* wrow0 = Wb + (n0 + lo) * WSR;
      const bf16_t* wrow1 = wrow0 + 16 * WSR;
      v8f c00 = {}, c01 = {}, c10 = {}, c11 = {};
#pragma unroll
      for (int k0 = 0; k0 < H; k0 += 32) {
        v16bf a0 = load_frag_a(hrow0 + k0 + hi * 8);
        v16bf a1 = load_frag_a(hrow1 + k0 + hi * 8);
        v16bf bb0 = load_frag_b(wrow0 + k0 + hi * 8);
        v16bf bb1 = load_frag_b(wrow1 + k0 + hi * 8);
        c00 = wmma_bf16(a0, bb0, c00);
        c01 = wmma_bf16(a0, bb1, c01);
        c10 = wmma_bf16(a1, bb0, c10);
        c11 = wmma_bf16(a1, bb1, c11);
      }
      bf16_t* t1base = t1b + bi * (H * TS);
      store_t1_tile(t1base, c00, n0,      m0,      lo, hi);
      store_t1_tile(t1base, c01, n0 + 16, m0,      lo, hi);
      store_t1_tile(t1base, c10, n0,      m0 + 16, lo, hi);
      store_t1_tile(t1base, c11, n0 + 16, m0 + 16, lo, hi);
    }
    __syncthreads();

    // GEMM2: h[node][ch] = relu( sum_k A[node][k]*T1^T[ch][k] + bias[ch] )
    // (M=96,N=128,K=96), same 2x2 blocking.
#pragma unroll
    for (int it = 0; it < 3; ++it) {
      int t  = it * 8 + wave;
      int bi = t / 12, r = t - bi * 12;
      int m0 = (r >> 2) * 32, n0 = (r & 3) * 32;
      const bf16_t* arow0 = Ab + (m0 + lo) * AS;
      const bf16_t* arow1 = arow0 + 16 * AS;
      const bf16_t* trow0 = t1b + bi * (H * TS) + (n0 + lo) * TS;
      const bf16_t* trow1 = trow0 + 16 * TS;
      v8f c00 = {}, c01 = {}, c10 = {}, c11 = {};
#pragma unroll
      for (int k0 = 0; k0 < NPAD; k0 += 32) {
        v16bf a0 = load_frag_a(arow0 + k0 + hi * 8);
        v16bf a1 = load_frag_a(arow1 + k0 + hi * 8);
        v16bf bb0 = load_frag_b(trow0 + k0 + hi * 8);
        v16bf bb1 = load_frag_b(trow1 + k0 + hi * 8);
        c00 = wmma_bf16(a0, bb0, c00);
        c01 = wmma_bf16(a0, bb1, c01);
        c10 = wmma_bf16(a1, bb0, c10);
        c11 = wmma_bf16(a1, bb1, c11);
      }
      float bias0 = biasb[n0 + lo];
      float bias1 = biasb[n0 + 16 + lo];
      bf16_t* hbase = hb + bi * (NPAD * HS);
      store_h_tile(hbase, c00, m0,      n0,      bias0, lo, hi);
      store_h_tile(hbase, c01, m0,      n0 + 16, bias1, lo, hi);
      store_h_tile(hbase, c10, m0 + 16, n0,      bias0, lo, hi);
      store_h_tile(hbase, c11, m0 + 16, n0 + 16, bias1, lo, hi);
    }
    __syncthreads();
  }

  // ---- output layer: logits = h @ W_out + b_out   (N = 9, padded to 16)
  for (int t = tid; t < 16 * H; t += 256) {
    int n = t / H, c = t - n * H;
    Wb[n * WSR + c] = (n < 9) ? (bf16_t)W_out[c * 9 + n] : (bf16_t)0.0f;
  }
  __syncthreads();

  for (int t = wave; t < BT * 6; t += 8) {
    int bi = t / 6, m0 = (t - bi * 6) * 16;
    const bf16_t* hrow = hb + bi * (NPAD * HS) + (m0 + lo) * HS;
    const bf16_t* wrow = Wb + lo * WSR;
    v8f acc = {};
#pragma unroll
    for (int k0 = 0; k0 < H; k0 += 32) {
      v16bf a = load_frag_a(hrow + k0 + hi * 8);
      v16bf b = load_frag_b(wrow + k0 + hi * 8);
      acc = wmma_bf16(a, b, acc);
    }
    if (lo < 9) {
      float bo = b_out[lo];
#pragma unroll
      for (int v = 0; v < 8; ++v) {
        int node = m0 + v + 8 * hi;
        if (node < NODES)
          out[((size_t)(b0 + bi) * NODES + node) * 9 + lo] = acc[v] + bo;
      }
    }
  }
}

extern "C" void kernel_launch(void* const* d_in, const int* in_sizes, int n_in,
                              void* d_out, int out_size, void* d_ws, size_t ws_size,
                              hipStream_t stream) {
  const int*   x        = (const int*)d_in[0];
  const float* W_in     = (const float*)d_in[1];
  const float* b_in     = (const float*)d_in[2];
  const float* W_layers = (const float*)d_in[3];
  const float* b_layers = (const float*)d_in[4];
  const float* W_out    = (const float*)d_in[5];
  const float* b_out    = (const float*)d_in[6];
  float* out = (float*)d_out;

  const int B = in_sizes[0] / NODES;          // 8192
  dim3 grid(B / BT), block(256);
  sudoku_gcn_fused<<<grid, block, SMEM_BYTES, stream>>>(
      x, W_in, b_in, W_layers, b_layers, W_out, b_out, out);
}
